// VPS_31628139168340
// MI455X (gfx1250) — matrix operand order repaired and checked by reading, not compiled, not run on hardware
//
#include <hip/hip_runtime.h>
#include <hip/hip_bf16.h>

#define BB     8
#define NPER   2048
#define NN     (BB * NPER)     // 16384
#define CD     256
#define KGE    16
#define KTOP   512
#define KNNK   32
#define EPSB   1e-5f

typedef __attribute__((ext_vector_type(8)))  float  v8f;
typedef __attribute__((ext_vector_type(16))) __bf16 v16bf;
typedef __attribute__((ext_vector_type(8)))  __bf16 v8bf;

// ---------------------------------------------------------------------------
// Kernel 0a: bf16 weight prep:  thb16 = bf16(theta_w), wqb16 = bf16(phi-theta)
//            bq = theta_b + phi_b
// ---------------------------------------------------------------------------
__global__ __launch_bounds__(256) void prep_weights_kernel(
    const float* __restrict__ thw, const float* __restrict__ phw,
    const float* __restrict__ thbias, const float* __restrict__ phbias,
    __bf16* __restrict__ thb16, __bf16* __restrict__ wqb16,
    float* __restrict__ bq) {
  const int i = blockIdx.x * 256 + threadIdx.x;   // 65536 total
  const float tv = thw[i];
  thb16[i] = (__bf16)tv;
  wqb16[i] = (__bf16)(phw[i] - tv);
  if (i < CD) bq[i] = thbias[i] + phbias[i];
}

// ---------------------------------------------------------------------------
// Kernel 0b: x -> bf16 (consumed 16x by the GEMM; halves streamed A bytes)
// ---------------------------------------------------------------------------
__global__ __launch_bounds__(256) void cvt_x_kernel(
    const float* __restrict__ in, __bf16* __restrict__ out) {
  const int i = blockIdx.x * 256 + threadIdx.x;   // NN*CD total
  out[i] = (__bf16)in[i];
}

// ---------------------------------------------------------------------------
// Kernel 1: T = x @ theta^T ; Q = x @ (phi-theta)^T + (theta_b+phi_b)
// One wave per 16x64 strip: 1 shared A fragment, 8 WMMAs per K step.
// ---------------------------------------------------------------------------
__global__ __launch_bounds__(256) void wmma_tq_kernel(
    const __bf16* __restrict__ xb, const __bf16* __restrict__ thb16,
    const __bf16* __restrict__ wqb16, const float* __restrict__ bq,
    float* __restrict__ T, float* __restrict__ Q) {
  const int wave = threadIdx.x >> 5;
  const int lane = threadIdx.x & 31;
  const int strip = blockIdx.x * 8 + wave;    // 1024*4 = 4096 strips
  const int mt = strip >> 2;                  // 0..1023  (16 rows)
  const int ns = strip & 3;                   // 0..3     (64 cols)
  const int m0 = mt * 16, n0 = ns * 64;
  const int hi = lane >> 4;                   // lane half (0/1)
  const int l15 = lane & 15;
  const int arow = m0 + l15;

  v8f accT[4] = {}; v8f accQ[4] = {};
  for (int k = 0; k < CD; k += 32) {
    // A fragment: 16x32 bf16, 16 contiguous K per lane half; two aligned 16B loads
    v16bf a;
    {
      const __bf16* p = xb + arow * CD + k + hi * 16;
      __builtin_prefetch(p + 32, 0, 0);       // global_prefetch_b8, next panel
      union { v16bf v; v8bf h[2]; } u;
      u.h[0] = *(const v8bf*)p;               // K = k + hi*16 + (0..7)
      u.h[1] = *(const v8bf*)(p + 8);         // K = k + hi*16 + 8 + (0..7)
      a = u.v;
    }
#pragma unroll
    for (int s = 0; s < 4; ++s) {
      // B fragments: 32x16 bf16; lane = col, 16 contiguous K per lane half
      const int wrow = n0 + s * 16 + l15;
      union { v16bf v; v8bf h[2]; } bt, bqf;
      const __bf16* pt = thb16 + wrow * CD + k + hi * 16;
      const __bf16* pq = wqb16 + wrow * CD + k + hi * 16;
      bt.h[0]  = *(const v8bf*)pt;   bt.h[1]  = *(const v8bf*)(pt + 8);
      bqf.h[0] = *(const v8bf*)pq;   bqf.h[1] = *(const v8bf*)(pq + 8);
      accT[s] = __builtin_amdgcn_wmma_f32_16x16x32_bf16(
          false, a, false, bt.v,  (short)0, accT[s], false, false);
      accQ[s] = __builtin_amdgcn_wmma_f32_16x16x32_bf16(
          false, a, false, bqf.v, (short)0, accQ[s], false, false);
    }
  }
  // C/D layout: VGPR r -> (M = m0 + r + hi*8, N = col)
#pragma unroll
  for (int s = 0; s < 4; ++s) {
    const int col = n0 + s * 16 + l15;
    const float biasq = bq[col];
#pragma unroll
    for (int r = 0; r < 8; ++r) {
      const int row = m0 + r + hi * 8;
      T[row * CD + col] = accT[s][r];
      Q[row * CD + col] = accQ[s][r] + biasq;
    }
  }
}

// ---------------------------------------------------------------------------
// Kernel 2: x1[i] = max_{j in N(i)} T[src_j] + Q[i] + x[i]
// Edges of node i are contiguous: [i*16, i*16+16). T is L2-resident (16 MB).
// ---------------------------------------------------------------------------
__global__ __launch_bounds__(256) void edge_max_kernel(
    const float* __restrict__ T, const float* __restrict__ Q,
    const float* __restrict__ x, const int* __restrict__ edge_src,
    float* __restrict__ x1) {
  const int node = blockIdx.x;
  const int c = threadIdx.x;
  const int* sp = edge_src + node * KGE;
  float m = -__builtin_inff();
#pragma unroll
  for (int j = 0; j < KGE; ++j) {
    const int s = sp[j];                      // uniform across block
    m = fmaxf(m, T[s * CD + c]);
  }
  x1[node * CD + c] = m + Q[node * CD + c] + x[node * CD + c];
}

// ---------------------------------------------------------------------------
// Kernel 3: per-node MLP 256->64->32->1 with eval-mode BN => score
// ---------------------------------------------------------------------------
__global__ __launch_bounds__(64) void mlp_score_kernel(
    const float* __restrict__ x1,
    const float* __restrict__ w1, const float* __restrict__ b1,
    const float* __restrict__ g1, const float* __restrict__ be1,
    const float* __restrict__ m1, const float* __restrict__ v1,
    const float* __restrict__ w2, const float* __restrict__ b2,
    const float* __restrict__ g2, const float* __restrict__ be2,
    const float* __restrict__ m2, const float* __restrict__ v2,
    const float* __restrict__ w3, const float* __restrict__ b3,
    float* __restrict__ score) {
  __shared__ float xr[CD];
  __shared__ float h1[64];
  __shared__ float h2[32];
  const int node = blockIdx.x;
  const int t = threadIdx.x;
  for (int i = t; i < CD; i += 64) xr[i] = x1[node * CD + i];
  __syncthreads();
  {
    float acc = b1[t];
    const float* wr = w1 + t * CD;
    for (int k = 0; k < CD; ++k) acc = fmaf(wr[k], xr[k], acc);
    acc = fmaxf(acc, 0.f);
    acc = (acc - m1[t]) * rsqrtf(v1[t] + EPSB) * g1[t] + be1[t];
    h1[t] = acc;
  }
  __syncthreads();
  if (t < 32) {
    float acc = b2[t];
    const float* wr = w2 + t * 64;
    for (int k = 0; k < 64; ++k) acc = fmaf(wr[k], h1[k], acc);
    acc = fmaxf(acc, 0.f);
    acc = (acc - m2[t]) * rsqrtf(v2[t] + EPSB) * g2[t] + be2[t];
    h2[t] = acc * w3[t];                      // fold final 1x32 matvec
  }
  __syncthreads();
  if (t == 0) {
    float s = b3[0];
    for (int k = 0; k < 32; ++k) s += h2[k];
    score[node] = s;
  }
}

// ---------------------------------------------------------------------------
// Kernel 4: per-batch descending bitonic sort of 2048 (score, idx) pairs,
// stable lower-index-first on ties (matches jax.lax.top_k).
// ---------------------------------------------------------------------------
__global__ __launch_bounds__(1024) void topk_kernel(
    const float* __restrict__ score, float* __restrict__ topk_out,
    int* __restrict__ perm, float* __restrict__ score_b_out) {
  __shared__ float v[NPER];
  __shared__ int   ix[NPER];
  const int b = blockIdx.x;
  const int t = threadIdx.x;
  for (int i = t; i < NPER; i += 1024) {
    const float s = score[b * NPER + i];
    v[i] = s; ix[i] = i;
    score_b_out[b * NPER + i] = s;
  }
  for (int ksz = 2; ksz <= NPER; ksz <<= 1) {
    for (int j = ksz >> 1; j > 0; j >>= 1) {
      __syncthreads();
      for (int i = t; i < NPER; i += 1024) {
        const int l = i ^ j;
        if (l > i) {
          const bool keep = (v[i] > v[l]) || (v[i] == v[l] && ix[i] < ix[l]);
          const bool dir = ((i & ksz) == 0);  // descending blocks
          if (keep != dir) {
            float tv = v[i]; v[i] = v[l]; v[l] = tv;
            int  ti = ix[i]; ix[i] = ix[l]; ix[l] = ti;
          }
        }
      }
    }
  }
  __syncthreads();
  if (t < KTOP) {
    topk_out[b * KTOP + t] = v[t];
    perm[b * KTOP + t] = b * NPER + ix[t];
  }
}

// ---------------------------------------------------------------------------
// Kernel 5: gather x_new / x_copy / xyz_sel via perm
// ---------------------------------------------------------------------------
__global__ __launch_bounds__(256) void gather_kernel(
    const int* __restrict__ perm, const float* __restrict__ x1,
    const float* __restrict__ x_origin, const float* __restrict__ xyz,
    float* __restrict__ x_new, float* __restrict__ x_copy,
    float* __restrict__ xyz_sel) {
  const int p = blockIdx.x;                   // 0..4095
  const int t = threadIdx.x;
  const int src = perm[p];
  x_new[p * CD + t]  = x1[src * CD + t];
  x_copy[p * CD + t] = x_origin[src * CD + t];
  if (t < 3) xyz_sel[p * 3 + t] = xyz[src * 3 + t];
}

// ---------------------------------------------------------------------------
// Kernel 6: 32-NN among 512 selected points per batch. Wave per query.
// Keys (f32bits<<32 | idx) extracted in increasing order => no mask storage.
// ---------------------------------------------------------------------------
__device__ inline unsigned long long wave_min_u64(unsigned long long val) {
#pragma unroll
  for (int off = 16; off > 0; off >>= 1) {
    const unsigned int lo = (unsigned int)val;
    const unsigned int hi = (unsigned int)(val >> 32);
    const unsigned int olo = __shfl_xor(lo, off, 32);
    const unsigned int ohi = __shfl_xor(hi, off, 32);
    const unsigned long long o = ((unsigned long long)ohi << 32) | olo;
    if (o < val) val = o;
  }
  return val;
}

__global__ __launch_bounds__(256) void knn_kernel(
    const float* __restrict__ xyz_sel, float* __restrict__ knn_out) {
  __shared__ float sx[KTOP], sy[KTOP], sz[KTOP];
  const int b = blockIdx.x >> 6;              // 64 blocks per batch
  const int wave = threadIdx.x >> 5;
  const int lane = threadIdx.x & 31;
  const int q = (blockIdx.x & 63) * 8 + wave; // 0..511
  for (int i = threadIdx.x; i < KTOP; i += 256) {
    const float* p = xyz_sel + (b * KTOP + i) * 3;
    sx[i] = p[0]; sy[i] = p[1]; sz[i] = p[2];
  }
  __syncthreads();
  const float qx = sx[q], qy = sy[q], qz = sz[q];
  unsigned long long kk[16];
#pragma unroll
  for (int c = 0; c < 16; ++c) {
    const int j = lane + c * 32;
    const float dx = sx[j] - qx, dy = sy[j] - qy, dz = sz[j] - qz;
    const float d2 = dx * dx + dy * dy + dz * dz;
    kk[c] = ((unsigned long long)__float_as_uint(d2) << 32) | (unsigned int)j;
  }
  unsigned long long thresh = 0ull;
  float* out = knn_out + (b * KTOP + q) * KNNK;
  for (int t = 0; t < KNNK; ++t) {
    unsigned long long best = ~0ull;
#pragma unroll
    for (int c = 0; c < 16; ++c)
      if (kk[c] >= thresh && kk[c] < best) best = kk[c];
    const unsigned long long w = wave_min_u64(best);
    thresh = w + 1ull;                        // keys unique (idx in low bits)
    if (lane == 0) out[t] = (float)(unsigned int)(w & 0xffffffffull);
  }
}

// ---------------------------------------------------------------------------
extern "C" void kernel_launch(void* const* d_in, const int* in_sizes, int n_in,
                              void* d_out, int out_size, void* d_ws, size_t ws_size,
                              hipStream_t stream) {
  const float* x        = (const float*)d_in[0];
  const float* x_origin = (const float*)d_in[1];
  const float* xyz      = (const float*)d_in[2];
  const int*   edge_src = (const int*)d_in[3];
  /* d_in[4] = edge_dst (implicit: contiguous segments) */
  const float* theta_w  = (const float*)d_in[5];
  const float* theta_b  = (const float*)d_in[6];
  const float* phi_w    = (const float*)d_in[7];
  const float* phi_b    = (const float*)d_in[8];
  const float* w1 = (const float*)d_in[9];   const float* b1  = (const float*)d_in[10];
  const float* g1 = (const float*)d_in[11];  const float* be1 = (const float*)d_in[12];
  const float* m1 = (const float*)d_in[13];  const float* v1  = (const float*)d_in[14];
  const float* w2 = (const float*)d_in[15];  const float* b2  = (const float*)d_in[16];
  const float* g2 = (const float*)d_in[17];  const float* be2 = (const float*)d_in[18];
  const float* m2 = (const float*)d_in[19];  const float* v2  = (const float*)d_in[20];
  const float* w3 = (const float*)d_in[21];  const float* b3  = (const float*)d_in[22];

  // workspace layout (f32/int first, then 16B-aligned bf16 arrays)
  float* T       = (float*)d_ws;                   // NN*CD f32
  float* Q       = T + (size_t)NN * CD;            // NN*CD f32
  float* X1      = Q + (size_t)NN * CD;            // NN*CD f32
  float* bq      = X1 + (size_t)NN * CD;           // CD f32
  float* scoreWS = bq + CD;                        // NN f32
  int*   perm    = (int*)(scoreWS + NN);           // BB*KTOP int
  float* xyz_sel = (float*)(perm + BB * KTOP);     // BB*KTOP*3 f32
  __bf16* xb     = (__bf16*)(xyz_sel + (size_t)BB * KTOP * 3); // NN*CD bf16
  __bf16* thb16  = xb + (size_t)NN * CD;           // CD*CD bf16
  __bf16* wqb16  = thb16 + (size_t)CD * CD;        // CD*CD bf16

  // output layout (tuple, flat, return order)
  float* out        = (float*)d_out;
  float* out_knn    = out;                                    // 8*512*32
  float* out_xnew   = out_knn  + (size_t)BB * KTOP * KNNK;    // 8*512*256
  float* out_xcopy  = out_xnew + (size_t)BB * KTOP * CD;      // 8*512*256
  float* out_scoreb = out_xcopy + (size_t)BB * KTOP * CD;     // 8*2048
  float* out_topk   = out_scoreb + NN;                        // 8*512

  prep_weights_kernel<<<(CD * CD) / 256, 256, 0, stream>>>(
      theta_w, phi_w, theta_b, phi_b, thb16, wqb16, bq);
  cvt_x_kernel<<<(NN * CD) / 256, 256, 0, stream>>>(x, xb);
  wmma_tq_kernel<<<(NN / 16) * (CD / 64) / 8, 256, 0, stream>>>(
      xb, thb16, wqb16, bq, T, Q);
  edge_max_kernel<<<NN, 256, 0, stream>>>(T, Q, x, edge_src, X1);
  mlp_score_kernel<<<NN, 64, 0, stream>>>(X1, w1, b1, g1, be1, m1, v1,
                                          w2, b2, g2, be2, m2, v2, w3, b3,
                                          scoreWS);
  topk_kernel<<<BB, 1024, 0, stream>>>(scoreWS, out_topk, perm, out_scoreb);
  gather_kernel<<<BB * KTOP, 256, 0, stream>>>(perm, X1, x_origin, xyz,
                                               out_xnew, out_xcopy, xyz_sel);
  knn_kernel<<<BB * 64, 256, 0, stream>>>(xyz_sel, out_knn);
}
